// QLSTMCell_65481071407962
// MI455X (gfx1250) — matrix operand order repaired and checked by reading, not compiled, not run on hardware
//
#include <hip/hip_runtime.h>
#include <math.h>

// Problem constants (match reference)
#define B_ROWS 16384
#define D_H    512
#define D_COMB 1024
#define M_TILE 32
#define KC     32

typedef __attribute__((ext_vector_type(16))) __bf16 v16bf;
typedef __attribute__((ext_vector_type(8)))  float  v8f;

union ABf {
    v16bf v;
    unsigned short us[16];
    uint4 u4[2];
};

struct QParams {
    const float* c;
    const float* b[4];    // gate biases [512]
    const float* W1[4];   // [512,16]
    const float* b1[4];   // [16]
    const float* W2[4];   // [16,1]
    const float* b2[4];   // [1]
    const unsigned short* wt;    // bf16 [4][512][1024] (gate, n, k)   in d_ws
    const unsigned short* comb;  // bf16 [16384][1024] concat(x,h)     in d_ws
    float* out_h;
    float* out_c;
};

__device__ __forceinline__ unsigned short f2bf(float f) {
    unsigned int u = __float_as_uint(f);
    unsigned int r = u + 0x7FFFu + ((u >> 16) & 1u);
    return (unsigned short)(r >> 16);
}

__device__ __forceinline__ unsigned int pk2(unsigned short a, unsigned short b) {
    return (unsigned int)a | ((unsigned int)b << 16);
}

// ---------------------------------------------------------------------------
// Kernel 0a: comb_bf16[row][k] = bf16( k<512 ? x[row][k] : h[row][k-512] )
// ---------------------------------------------------------------------------
__global__ void comb_prep(const float* __restrict__ x, const float* __restrict__ h,
                          unsigned short* __restrict__ comb) {
    int t   = blockIdx.x * 256 + threadIdx.x;   // 16384*128 threads
    int row = t >> 7;
    int kb  = (t & 127) * 8;
    const float* src = (kb < 512) ? (x + (size_t)row * 512 + kb)
                                  : (h + (size_t)row * 512 + (kb - 512));
    float4 f0 = ((const float4*)src)[0];
    float4 f1 = ((const float4*)src)[1];
    uint4 o;
    o.x = pk2(f2bf(f0.x), f2bf(f0.y));
    o.y = pk2(f2bf(f0.z), f2bf(f0.w));
    o.z = pk2(f2bf(f1.x), f2bf(f1.y));
    o.w = pk2(f2bf(f1.z), f2bf(f1.w));
    *(uint4*)(comb + (size_t)row * 1024 + kb) = o;
}

// ---------------------------------------------------------------------------
// Kernel 0b: transpose + convert gate weights: wt[g][n][k] = bf16(W_g[k][n])
// ---------------------------------------------------------------------------
__global__ void wt_prep(const float* __restrict__ Wf, const float* __restrict__ Wi,
                        const float* __restrict__ Wu, const float* __restrict__ Wo,
                        unsigned short* __restrict__ wt) {
    int t = blockIdx.x * 256 + threadIdx.x;   // 4*512*1024 total
    int g   = t >> 19;
    int rem = t & 524287;
    int n   = rem >> 10;
    int k   = rem & 1023;
    const float* W = (g == 0) ? Wf : (g == 1) ? Wi : (g == 2) ? Wu : Wo;
    wt[t] = f2bf(W[k * 512 + n]);
}

// ---------------------------------------------------------------------------
// Kernel 1: fused 4-gate GEMM (bf16 WMMA, f32 accum) + qgate MLP + LSTM update.
// Workgroup: 512 threads (16 waves), 32 rows, all 512 cols, all 4 gates.
// Per wave: 1 gate x 8 n-tiles x 2 m-subtiles = 16 acc tiles (128 VGPRs).
// ---------------------------------------------------------------------------
__global__ __launch_bounds__(512) void qlstm_main(QParams P) {
    __shared__ float zs[16][512];        // 32 KB: one (subtile, gate) z tile
    __shared__ float part[16][16][2];    // 2 KB: qgate partial sums
    __shared__ float gatesc[4][32];      // gate scalars per row

    const int t    = threadIdx.x;
    const int w    = t >> 5;         // wave 0..15
    const int lane = t & 31;
    const int lo   = lane & 15;
    const int hi   = lane >> 4;      // 0 / 1
    const int g    = w >> 2;         // this wave's gate (0..3)
    const int wq   = w & 3;          // wave's quarter of the N range
    const int m0   = blockIdx.x * M_TILE;

    // Single A base pointer; subtile delta ms*16*1024 folds into load offset.
    const unsigned short* apt =
        P.comb + ((size_t)(m0 + lo) * 1024) + hi * 8;
    // Single B base pointer; per-pair delta q*16*1024 folds into load offset.
    const unsigned short* wb =
        P.wt + (((size_t)g * 512 + wq * 128 + lo) * 1024) + hi * 8;

    float biasv[8];
#pragma unroll
    for (int q = 0; q < 8; ++q)
        biasv[q] = P.b[g][wq * 128 + q * 16 + lo];

    v8f acc0[8], acc1[8];
#pragma unroll
    for (int q = 0; q < 8; ++q) {
        acc0[q] = (v8f){0.f, 0.f, 0.f, 0.f, 0.f, 0.f, 0.f, 0.f};
        acc1[q] = (v8f){0.f, 0.f, 0.f, 0.f, 0.f, 0.f, 0.f, 0.f};
    }

    // ---------------- main K loop: K = 1024 in chunks of 32 ----------------
#pragma unroll 1
    for (int k0 = 0; k0 < D_COMB; k0 += KC) {
        // Two A fragments (16x32 bf16, ISA layout; loads are uint4 pairs)
        const uint4* pa0 = (const uint4*)(apt + k0);
        const uint4* pa1 = (const uint4*)(apt + k0 + 16384);
        ABf af0, af1;
        af0.u4[0] = pa0[0];
        af0.u4[1] = pa0[2];     // +16 bf16 elements
        af1.u4[0] = pa1[0];
        af1.u4[1] = pa1[2];
        if (k0 + KC < D_COMB) {
            __builtin_prefetch(wb + k0 + KC, 0, 1);
            __builtin_prefetch(apt + k0 + KC, 0, 1);
        }

        // Per n-tile: load one B fragment, immediately use it twice.
#pragma unroll
        for (int q = 0; q < 8; ++q) {
            const uint4* pb = (const uint4*)(wb + k0 + q * 16384);
            ABf bq;
            bq.u4[0] = pb[0];
            bq.u4[1] = pb[2];
            acc0[q] = __builtin_amdgcn_wmma_f32_16x16x32_bf16(
                false, af0.v, false, bq.v, (short)0, acc0[q], false, false);
            acc1[q] = __builtin_amdgcn_wmma_f32_16x16x32_bf16(
                false, af1.v, false, bq.v, (short)0, acc1[q], false, false);
        }
    }

    // ---------------- qgate MLP per (subtile, gate) ----------------
    const int qn = t >> 8;         // n-half 0..1
    const int rr = (t >> 4) & 15;  // row within subtile
    const int jj = t & 15;         // hidden unit

#pragma unroll
    for (int ms = 0; ms < 2; ++ms) {
#pragma unroll 1
        for (int gg = 0; gg < 4; ++gg) {
            __syncthreads();   // protect zs/part from previous iteration
            if (g == gg) {
                // dump this wave's z tiles for subtile ms into LDS (+bias)
#pragma unroll
                for (int q = 0; q < 8; ++q) {
                    int col = wq * 128 + q * 16 + lo;
#pragma unroll
                    for (int v = 0; v < 8; ++v) {
                        int r = v + hi * 8;   // C layout: VGPR v -> row v+8*hi
                        float zv = (ms == 0) ? acc0[q][v] : acc1[q][v];
                        zs[r][col] = zv + biasv[q];
                    }
                }
            }
            __syncthreads();

            // hidden_j(r) partials: z[r,:] @ W1[:,j], split over 2 n-halves
            const float* W1 = P.W1[gg];
            float s    = 0.f;
            int   nbeg = qn * 256;
#pragma unroll 8
            for (int n = 0; n < 256; ++n)
                s += zs[rr][nbeg + n] * W1[(nbeg + n) * 16 + jj];
            part[rr][jj][qn] = s;
            __syncthreads();

            if (t < 256) {
                float hsum = part[rr][jj][0] + part[rr][jj][1] + P.b1[gg][jj];
                float a  = fmaxf(hsum, 0.f);
                float pj = a * P.W2[gg][jj];
                pj += __shfl_xor(pj, 1, 16);
                pj += __shfl_xor(pj, 2, 16);
                pj += __shfl_xor(pj, 4, 16);
                pj += __shfl_xor(pj, 8, 16);
                if (jj == 0) {
                    float sv    = pj + P.b2[gg][0];
                    float inner = tanhf(sv);
                    float gv = (gg == 2) ? tanhf(inner)
                                         : 1.f / (1.f + __expf(-inner));
                    gatesc[gg][ms * 16 + rr] = gv;
                }
            }
        }
    }
    __syncthreads();

    // ---------------- fused LSTM elementwise update ----------------
    // 32 rows x 512 cols = 4096 float4; 8 per thread
#pragma unroll
    for (int it = 0; it < 8; ++it) {
        int e  = t + it * 512;
        int rl = e >> 7;        // local row 0..31
        int ci = e & 127;       // float4 index within row
        int row = m0 + rl;
        float f_ = gatesc[0][rl];
        float i_ = gatesc[1][rl];
        float u_ = gatesc[2][rl];
        float o_ = gatesc[3][rl];
        float4 c4 = ((const float4*)(P.c + (size_t)row * 512))[ci];
        float4 cn, hn;
        cn.x = f_ * c4.x + i_ * u_;
        cn.y = f_ * c4.y + i_ * u_;
        cn.z = f_ * c4.z + i_ * u_;
        cn.w = f_ * c4.w + i_ * u_;
        hn.x = o_ * tanhf(cn.x);
        hn.y = o_ * tanhf(cn.y);
        hn.z = o_ * tanhf(cn.z);
        hn.w = o_ * tanhf(cn.w);
        ((float4*)(P.out_c + (size_t)row * 512))[ci] = cn;
        ((float4*)(P.out_h + (size_t)row * 512))[ci] = hn;
    }
}

extern "C" void kernel_launch(void* const* d_in, const int* in_sizes, int n_in,
                              void* d_out, int out_size, void* d_ws, size_t ws_size,
                              hipStream_t stream) {
    (void)in_sizes; (void)n_in; (void)out_size; (void)ws_size;

    const float* x = (const float*)d_in[0];
    const float* h = (const float*)d_in[1];

    // d_ws layout: [0, 4MB) bf16 weights ; [4MB, 36MB) bf16 comb
    unsigned short* wt   = (unsigned short*)d_ws;
    unsigned short* comb = wt + (size_t)4 * 512 * 1024;

    QParams P;
    P.c = (const float*)d_in[2];
    for (int g = 0; g < 4; ++g) {
        int b0 = 3 + 6 * g;          // W_n, b_n, gn_W1, gn_b1, gn_W2, gn_b2
        P.b[g]  = (const float*)d_in[b0 + 1];
        P.W1[g] = (const float*)d_in[b0 + 2];
        P.b1[g] = (const float*)d_in[b0 + 3];
        P.W2[g] = (const float*)d_in[b0 + 4];
        P.b2[g] = (const float*)d_in[b0 + 5];
    }
    P.wt    = wt;
    P.comb  = comb;
    P.out_h = (float*)d_out;
    P.out_c = (float*)d_out + (size_t)B_ROWS * D_H;

    wt_prep<<<(4 * 512 * 1024) / 256, 256, 0, stream>>>(
        (const float*)d_in[3], (const float*)d_in[9],
        (const float*)d_in[15], (const float*)d_in[21], wt);

    comb_prep<<<(B_ROWS * 128) / 256, 256, 0, stream>>>(x, h, comb);

    qlstm_main<<<B_ROWS / M_TILE, 512, 0, stream>>>(P);
}
